// PT_SSL_Model_6794638262324
// MI455X (gfx1250) — matrix operand order repaired and checked by reading, not compiled, not run on hardware
//
#include <hip/hip_runtime.h>
#include <hip/hip_bf16.h>
#include <cstdint>
#include <cstddef>

// ---------------------------------------------------------------------------
// PT_SSL transformer forward on gfx1250 (MI455X), bf16 WMMA everywhere.
// GEMMs: 128x128 block tile, 8 waves, 32x64 wave tile (2x4 WMMA / K-step),
// double-buffered LDS with register prefetch; weights pre-transposed so every
// WMMA fragment is exactly two ds_load_b128.
// ---------------------------------------------------------------------------

typedef __bf16 bf16;
typedef __bf16 v16bf __attribute__((ext_vector_type(16)));
typedef __bf16 v8bf  __attribute__((ext_vector_type(8)));
typedef float  v8f   __attribute__((ext_vector_type(8)));

static constexpr int kB   = 1024;
static constexpr int kC   = 3;
static constexpr int kNP  = 10;
static constexpr int kPL  = 100;
static constexpr int kD   = 768;
static constexpr int kH   = 12;
static constexpr int kL   = 12;
static constexpr int kN   = 12;      // NP + 2 tokens
static constexpr int kHD  = 64;
static constexpr int kSF  = 56;
static constexpr int kTF  = 24;
static constexpr int kKP  = 320;     // padded patch K (300 -> 320)
static constexpr int kNPAD= 384;     // padded head N (300 -> 384, mult of 128)
static constexpr int kRowsTok = kB * kN;    // 12288
static constexpr int kRowsPat = kB * kNP;   // 10240

__device__ __forceinline__ v16bf cat16(v8bf lo, v8bf hi) {
  return __builtin_shufflevector(lo, hi, 0, 1, 2, 3, 4, 5, 6, 7,
                                 8, 9, 10, 11, 12, 13, 14, 15);
}

// ---------------------------------------------------------------------------
// tiled transpose + fp32->bf16 + zero-pad:  out[Cp][Rp] <- in[R0][C0]^T
// ---------------------------------------------------------------------------
__global__ __launch_bounds__(256) void k_transpose_cvt(
    bf16* __restrict__ out, const float* __restrict__ in,
    int Rp, int Cp, int R0, int C0,
    long long inLayerStride, long long outLayerStride) {
  __shared__ float tile[32][33];
  in  += (long long)blockIdx.z * inLayerStride;
  out += (long long)blockIdx.z * outLayerStride;
  const int rt = blockIdx.x * 32;   // source-row tile (= out col / K dim)
  const int ct = blockIdx.y * 32;   // source-col tile (= out row / N dim)
  const int tx = threadIdx.x & 31, ty = threadIdx.x >> 5;
#pragma unroll
  for (int j = 0; j < 4; ++j) {
    int r = rt + ty + j * 8, c = ct + tx;
    tile[ty + j * 8][tx] = (r < R0 && c < C0) ? in[(long long)r * C0 + c] : 0.0f;
  }
  __syncthreads();
#pragma unroll
  for (int j = 0; j < 4; ++j) {
    int n = ct + ty + j * 8, k = rt + tx;
    out[(long long)n * Rp + k] = (bf16)tile[tx][ty + j * 8];
  }
}

__global__ __launch_bounds__(256) void k_pad_bias(float* __restrict__ out,
                                                  const float* __restrict__ in,
                                                  int Np, int N0) {
  int i = blockIdx.x * 256 + threadIdx.x;
  if (i < Np) out[i] = (i < N0) ? in[i] : 0.0f;
}

// flat_raw (padded bf16): out[b*NP+p][c] = patches[b][c/100][p][c%100], 320-wide
__global__ __launch_bounds__(256) void k_prep_patches(bf16* __restrict__ out,
                                                      const float* __restrict__ patches) {
  long long i = (long long)blockIdx.x * 256 + threadIdx.x;
  long long n = (long long)kRowsPat * kKP;
  if (i >= n) return;
  int row = (int)(i / kKP), c = (int)(i % kKP);
  int b = row / kNP, p = row % kNP;
  float v = 0.0f;
  if (c < kC * kPL) {
    int ch = c / kPL, l = c % kPL;
    v = patches[(((long long)b * kC + ch) * kNP + p) * kPL + l];
  }
  out[i] = (bf16)v;
}

__global__ __launch_bounds__(1) void k_zero2(float* a) { a[0] = 0.0f; a[1] = 0.0f; }

__global__ __launch_bounds__(1) void k_final(float* out, const float* acc) {
  out[0] = acc[0] / acc[1];
}

// ---------------------------------------------------------------------------
// WMMA GEMM:  out[M,N] = A[M,K](bf16) @ W[K,N] with W supplied TRANSPOSED as
// WT[N][K] (bf16).  (+bias) (relu) (+resid).  M%128==0, N%128==0, K%32==0.
// Block 256 thr = 8 waves; block tile 128x128; wave tile 32x64 (2x4 WMMA).
// Double-buffered LDS, register-prefetch of the next K-tile.
// ---------------------------------------------------------------------------
static constexpr int GBM = 128, GBN = 128, GBK = 32, GPAD = 8;

template <bool RELU, bool RES, bool OUTF, bool OUTB>
__global__ __launch_bounds__(256) void k_gemm_bf16(
    const bf16* __restrict__ A, const bf16* __restrict__ WT,
    const float* __restrict__ bias, const float* __restrict__ resid,
    float* __restrict__ outF, bf16* __restrict__ outB,
    int M, int K, int Nn) {
  __shared__ bf16 sA[2][GBM][GBK + GPAD];
  __shared__ bf16 sB[2][GBN][GBK + GPAD];

  const int bm = blockIdx.y * GBM;
  const int bn = blockIdx.x * GBN;
  const int tid  = threadIdx.x;
  const int wave = tid >> 5;
  const int lane = tid & 31;
  const int wm = (wave >> 1) * 32;      // 0,32,64,96
  const int wn = (wave & 1) * 64;       // 0,64
  const int m16 = lane & 15;
  const int kh  = lane >> 4;            // lane half

  const int srow = tid >> 2;            // 0..63
  const int scol = (tid & 3) * 8;       // 0,8,16,24

  uint4 ra0, ra1, rb0, rb1;
  auto fetch = [&](int k0) {
    ra0 = *(const uint4*)(A  + (size_t)(bm + srow)      * K + k0 + scol);
    ra1 = *(const uint4*)(A  + (size_t)(bm + 64 + srow) * K + k0 + scol);
    rb0 = *(const uint4*)(WT + (size_t)(bn + srow)      * K + k0 + scol);
    rb1 = *(const uint4*)(WT + (size_t)(bn + 64 + srow) * K + k0 + scol);
  };
  auto stage = [&](int buf) {
    *(uint4*)&sA[buf][srow][scol]      = ra0;
    *(uint4*)&sA[buf][64 + srow][scol] = ra1;
    *(uint4*)&sB[buf][srow][scol]      = rb0;
    *(uint4*)&sB[buf][64 + srow][scol] = rb1;
  };

  v8f acc[2][4];
#pragma unroll
  for (int i = 0; i < 2; ++i)
#pragma unroll
    for (int j = 0; j < 4; ++j) acc[i][j] = {};

  fetch(0);
  stage(0);
  __syncthreads();

  int buf = 0;
  for (int k0 = 0; k0 < K; k0 += GBK) {
    const bool pf = (k0 + GBK) < K;
    if (pf) fetch(k0 + GBK);

    // fragments: each = two 16B LDS loads (lane half h: k = 8h..8h+7, 16+8h..16+8h+7)
    v16bf af[2], bfr[4];
#pragma unroll
    for (int i = 0; i < 2; ++i) {
      const bf16* ap = &sA[buf][wm + i * 16 + m16][kh * 8];
      af[i] = cat16(*(const v8bf*)ap, *(const v8bf*)(ap + 16));
    }
#pragma unroll
    for (int j = 0; j < 4; ++j) {
      const bf16* bp = &sB[buf][wn + j * 16 + m16][kh * 8];
      bfr[j] = cat16(*(const v8bf*)bp, *(const v8bf*)(bp + 16));
    }
#pragma unroll
    for (int i = 0; i < 2; ++i)
#pragma unroll
      for (int j = 0; j < 4; ++j)
        acc[i][j] = __builtin_amdgcn_wmma_f32_16x16x32_bf16(
            false, af[i], false, bfr[j], (short)0, acc[i][j], false, false);

    if (pf) stage(buf ^ 1);
    __syncthreads();
    buf ^= 1;
  }

  // epilogue: C/D layout -> row = base + r + 8*(lane>=16), col = base + lane%16
  const int colb = bn + wn + m16;
  const int rowb = bm + wm + kh * 8;
#pragma unroll
  for (int i = 0; i < 2; ++i) {
#pragma unroll
    for (int j = 0; j < 4; ++j) {
      const int col = colb + j * 16;
      const float bv = bias[col];
#pragma unroll
      for (int r = 0; r < 8; ++r) {
        const int row = rowb + i * 16 + r;
        float v = acc[i][j][r] + bv;
        if (RELU) v = fmaxf(v, 0.0f);
        if (RES)  v += resid[(size_t)row * Nn + col];
        if (OUTF) outF[(size_t)row * Nn + col] = v;
        if (OUTB) outB[(size_t)row * Nn + col] = (bf16)v;
      }
    }
  }
}

// ---------------------------------------------------------------------------
// embed raw tokens: x[b*12+n][d]
// ---------------------------------------------------------------------------
__global__ __launch_bounds__(256) void k_embed(
    float* __restrict__ x, const float* __restrict__ pv,
    const float* __restrict__ stats, const float* __restrict__ topo,
    const float* __restrict__ stat_w, const float* __restrict__ stat_b,
    const float* __restrict__ topo_w, const float* __restrict__ topo_b,
    const unsigned char* __restrict__ mask, const float* __restrict__ mask_token) {
  int row = blockIdx.x;
  int b = row / kN, n = row % kN;
#pragma unroll
  for (int j = 0; j < 3; ++j) {
    int d = threadIdx.x + j * 256;
    float val;
    if (n < kNP) {
      val = mask[b * kNP + n] ? mask_token[d]
                              : pv[((size_t)b * kNP + n) * kD + d];
    } else if (n == kNP) {
      float a = stat_b[d];
      for (int i = 0; i < kSF; ++i) a += stats[b * kSF + i] * stat_w[i * kD + d];
      val = a;
    } else {
      float a = topo_b[d];
      for (int i = 0; i < kTF; ++i) a += topo[b * kTF + i] * topo_w[i * kD + d];
      val = a;
    }
    x[(size_t)row * kD + d] = val;
  }
}

// ---------------------------------------------------------------------------
// LayerNorm over 768, one wave per row. outF and/or outB (bf16) optional.
// ---------------------------------------------------------------------------
__global__ __launch_bounds__(32) void k_ln(const float* __restrict__ x,
                                           const float* __restrict__ g,
                                           const float* __restrict__ be,
                                           float* __restrict__ outF,
                                           bf16* __restrict__ outB) {
  const int row  = blockIdx.x;
  const int lane = threadIdx.x;
  const float* xr = x + (size_t)row * kD;
  float vals[24];
  float s = 0.0f;
#pragma unroll
  for (int i = 0; i < 24; ++i) { vals[i] = xr[lane + i * 32]; s += vals[i]; }
#pragma unroll
  for (int m = 16; m; m >>= 1) s += __shfl_xor(s, m, 32);
  const float mean = s * (1.0f / kD);
  float vs = 0.0f;
#pragma unroll
  for (int i = 0; i < 24; ++i) { float d = vals[i] - mean; vs += d * d; }
#pragma unroll
  for (int m = 16; m; m >>= 1) vs += __shfl_xor(vs, m, 32);
  const float inv = rsqrtf(vs * (1.0f / kD) + 1e-5f);
#pragma unroll
  for (int i = 0; i < 24; ++i) {
    int d = lane + i * 32;
    float y = (vals[i] - mean) * inv * g[d] + be[d];
    if (outF) outF[(size_t)row * kD + d] = y;
    if (outB) outB[(size_t)row * kD + d] = (bf16)y;
  }
}

// ---------------------------------------------------------------------------
// Attention: one wave per (b, head). RoPE on q/k, 12x12 softmax, out (bf16).
// ---------------------------------------------------------------------------
__global__ __launch_bounds__(128) void k_attn(const bf16* __restrict__ qkv,
                                              bf16* __restrict__ outB) {
  __shared__ float sm[4][3 * kN * kHD + kN * kN];
  const int wave = threadIdx.x >> 5;
  const int lane = threadIdx.x & 31;
  const int gid  = blockIdx.x * 4 + wave;   // b*H + h
  const int b = gid / kH, h = gid % kH;

  float* q  = sm[wave];
  float* kk = q + kN * kHD;
  float* v  = kk + kN * kHD;
  float* sc = v + kN * kHD;

  const bf16* base = qkv + (size_t)b * kN * (3 * kD) + (size_t)h * kHD;

  for (int p = lane; p < kN * 32; p += 32) {
    int n = p >> 5, i = p & 31;
    float ang = (float)n * __powf(10000.0f, -(2.0f * i) / (float)kHD);
    float c, s;
    __sincosf(ang, &s, &c);
    const bf16* rq = base + (size_t)n * (3 * kD);
    const bf16* rk = rq + kD;
    const bf16* rv = rq + 2 * kD;
    float xr = (float)rq[2 * i], xi = (float)rq[2 * i + 1];
    q[n * kHD + 2 * i]     = xr * c - xi * s;
    q[n * kHD + 2 * i + 1] = xr * s + xi * c;
    xr = (float)rk[2 * i]; xi = (float)rk[2 * i + 1];
    kk[n * kHD + 2 * i]     = xr * c - xi * s;
    kk[n * kHD + 2 * i + 1] = xr * s + xi * c;
    v[n * kHD + 2 * i]     = (float)rv[2 * i];
    v[n * kHD + 2 * i + 1] = (float)rv[2 * i + 1];
  }
  __syncthreads();

  const float scale = 0.125f;  // 1/sqrt(64)
  for (int p = lane; p < kN * kN; p += 32) {
    int i = p / kN, j = p % kN;
    float a = 0.0f;
#pragma unroll
    for (int d = 0; d < kHD; ++d) a += q[i * kHD + d] * kk[j * kHD + d];
    sc[p] = a * scale;
  }
  __syncthreads();

  if (lane < kN) {
    float mx = -1e30f;
#pragma unroll
    for (int j = 0; j < kN; ++j) mx = fmaxf(mx, sc[lane * kN + j]);
    float e[kN], sum = 0.0f;
#pragma unroll
    for (int j = 0; j < kN; ++j) { e[j] = __expf(sc[lane * kN + j] - mx); sum += e[j]; }
    float invs = 1.0f / sum;
#pragma unroll
    for (int j = 0; j < kN; ++j) sc[lane * kN + j] = e[j] * invs;
  }
  __syncthreads();

  for (int p = lane; p < kN * kHD; p += 32) {
    int i = p >> 6, d = p & 63;
    float a = 0.0f;
#pragma unroll
    for (int j = 0; j < kN; ++j) a += sc[i * kN + j] * v[j * kHD + d];
    outB[((size_t)(b * kN + i)) * kD + h * kHD + d] = (bf16)a;
  }
}

// ---------------------------------------------------------------------------
// gather token rows n<10 -> bf16 head input [B*NP, D]
// ---------------------------------------------------------------------------
__global__ __launch_bounds__(256) void k_gather(bf16* __restrict__ out,
                                                const float* __restrict__ x) {
  long long i = (long long)blockIdx.x * 256 + threadIdx.x;
  long long n = (long long)kRowsPat * kD;
  if (i >= n) return;
  int row = (int)(i / kD), d = (int)(i % kD);
  int b = row / kNP, p = row % kNP;
  out[i] = (bf16)x[((size_t)b * kN + p) * kD + d];
}

// ---------------------------------------------------------------------------
// masked MSE accumulation (one wave per (b,p))
// ---------------------------------------------------------------------------
__global__ __launch_bounds__(32) void k_loss(const float* __restrict__ pred,
                                             const float* __restrict__ patches,
                                             const unsigned char* __restrict__ mask,
                                             float* __restrict__ acc) {
  const int row = blockIdx.x;
  if (!mask[row]) return;
  const int b = row / kNP, p = row % kNP;
  const int lane = threadIdx.x;
  float s = 0.0f;
  for (int c = lane; c < kC * kPL; c += 32) {
    int ch = c / kPL, l = c % kPL;
    float t = patches[(((long long)b * kC + ch) * kNP + p) * kPL + l];
    float d = pred[(size_t)row * kNPAD + c] - t;
    s += d * d;
  }
#pragma unroll
  for (int m = 16; m; m >>= 1) s += __shfl_xor(s, m, 32);
  if (lane == 0) {
    atomicAdd(&acc[0], s);
    atomicAdd(&acc[1], (float)(kC * kPL));
  }
}

// ---------------------------------------------------------------------------
// launch
// ---------------------------------------------------------------------------
enum GemmMode { G_OUTF = 0, G_OUTB = 1, G_RES_OUTF = 2, G_RELU_OUTB = 3 };

static inline void gemm(hipStream_t st, GemmMode mode,
                        const bf16* A, const bf16* WT, const float* bias,
                        const float* resid, float* outF, bf16* outB,
                        int M, int K, int Nn) {
  dim3 grid(Nn / GBN, M / GBM);
  switch (mode) {
    case G_OUTF:
      k_gemm_bf16<false, false, true, false><<<grid, 256, 0, st>>>(
          A, WT, bias, nullptr, outF, nullptr, M, K, Nn);
      break;
    case G_OUTB:
      k_gemm_bf16<false, false, false, true><<<grid, 256, 0, st>>>(
          A, WT, bias, nullptr, nullptr, outB, M, K, Nn);
      break;
    case G_RES_OUTF:
      k_gemm_bf16<false, true, true, false><<<grid, 256, 0, st>>>(
          A, WT, bias, resid, outF, nullptr, M, K, Nn);
      break;
    case G_RELU_OUTB:
      k_gemm_bf16<true, false, false, true><<<grid, 256, 0, st>>>(
          A, WT, bias, nullptr, nullptr, outB, M, K, Nn);
      break;
  }
}

extern "C" void kernel_launch(void* const* d_in, const int* in_sizes, int n_in,
                              void* d_out, int out_size, void* d_ws, size_t ws_size,
                              hipStream_t stream) {
  const float* patches   = (const float*)d_in[0];
  const float* stats     = (const float*)d_in[1];
  const float* topo      = (const float*)d_in[2];
  const unsigned char* mask = (const unsigned char*)d_in[3];
  const float* patch_w   = (const float*)d_in[4];
  const float* patch_b   = (const float*)d_in[5];
  const float* stat_w    = (const float*)d_in[6];
  const float* stat_b    = (const float*)d_in[7];
  const float* topo_w    = (const float*)d_in[8];
  const float* topo_b    = (const float*)d_in[9];
  const float* mask_tok  = (const float*)d_in[10];
  const float* emb_g     = (const float*)d_in[11];
  const float* emb_b     = (const float*)d_in[12];
  const float* ln1_g     = (const float*)d_in[13];
  const float* ln1_b     = (const float*)d_in[14];
  const float* qkv_w     = (const float*)d_in[15];
  const float* qkv_b     = (const float*)d_in[16];
  const float* proj_w    = (const float*)d_in[17];
  const float* proj_b    = (const float*)d_in[18];
  const float* ln2_g     = (const float*)d_in[19];
  const float* ln2_b     = (const float*)d_in[20];
  const float* f1_w      = (const float*)d_in[21];
  const float* f1_b      = (const float*)d_in[22];
  const float* f2_w      = (const float*)d_in[23];
  const float* f2_b      = (const float*)d_in[24];
  const float* h1_w      = (const float*)d_in[25];
  const float* h1_b      = (const float*)d_in[26];
  const float* h2_w      = (const float*)d_in[27];
  const float* h2_b      = (const float*)d_in[28];

  // ---- workspace carve-up ----
  char* w = (char*)d_ws;
  size_t off = 0;
  auto alloc = [&](size_t bytes) -> void* {
    void* p = w + off;
    off = (off + bytes + 255) & ~(size_t)255;
    return p;
  };
  // transposed bf16 weights (WT[N][K])
  bf16* bApat  = (bf16*)alloc((size_t)kRowsPat * kKP * 2);
  bf16* bWpat  = (bf16*)alloc((size_t)kD * kKP * 2);             // [768][320]
  bf16* bWqkv  = (bf16*)alloc((size_t)kL * 3 * kD * kD * 2);     // [2304][768] x12
  bf16* bWproj = (bf16*)alloc((size_t)kL * kD * kD * 2);         // [768][768] x12
  bf16* bWf1   = (bf16*)alloc((size_t)kL * 2 * kD * kD * 2);     // [1536][768] x12
  bf16* bWf2   = (bf16*)alloc((size_t)kL * kD * 2 * kD * 2);     // [768][1536] x12
  bf16* bWh1   = (bf16*)alloc((size_t)kD * kD * 2);              // [768][768]
  bf16* bWh2   = (bf16*)alloc((size_t)kNPAD * kD * 2);           // [384][768]
  bf16* bH     = (bf16*)alloc((size_t)kRowsTok * kD * 2);        // LN out / head in
  bf16* bQKV   = (bf16*)alloc((size_t)kRowsTok * 3 * kD * 2);    // also FFN mid
  bf16* bAttn  = (bf16*)alloc((size_t)kRowsTok * kD * 2);        // also head mid
  float* fPV   = (float*)alloc((size_t)kRowsPat * kD * 4);       // also pred
  float* fX    = (float*)alloc((size_t)kRowsTok * kD * 4);
  float* fBias2= (float*)alloc(kNPAD * 4);
  float* fAcc  = (float*)alloc(256);
  (void)ws_size; (void)n_in; (void)in_sizes; (void)out_size;

  bf16* bMid   = bQKV;   // FFN intermediate [12288,1536] aliases qkv buffer
  bf16* bHeadI = bH;     // head input [10240,768]
  bf16* bHeadM = bAttn;  // head mid   [10240,768]
  float* fPred = fPV;    // pred [10240,384]

  auto blocks = [](long long n) { return (unsigned)((n + 255) / 256); };

  k_zero2<<<1, 1, 0, stream>>>(fAcc);

  // ---- weight conversion: fp32 [K][N] -> bf16 WT[N][K] (padded) ----
  k_transpose_cvt<<<dim3(kKP / 32, kD / 32, 1), 256, 0, stream>>>(
      bWpat, patch_w, kKP, kD, kC * kPL, kD, 0, 0);
  k_transpose_cvt<<<dim3(kD / 32, (3 * kD) / 32, kL), 256, 0, stream>>>(
      bWqkv, qkv_w, kD, 3 * kD, kD, 3 * kD,
      (long long)kD * 3 * kD, (long long)3 * kD * kD);
  k_transpose_cvt<<<dim3(kD / 32, kD / 32, kL), 256, 0, stream>>>(
      bWproj, proj_w, kD, kD, kD, kD,
      (long long)kD * kD, (long long)kD * kD);
  k_transpose_cvt<<<dim3(kD / 32, (2 * kD) / 32, kL), 256, 0, stream>>>(
      bWf1, f1_w, kD, 2 * kD, kD, 2 * kD,
      (long long)kD * 2 * kD, (long long)2 * kD * kD);
  k_transpose_cvt<<<dim3((2 * kD) / 32, kD / 32, kL), 256, 0, stream>>>(
      bWf2, f2_w, 2 * kD, kD, 2 * kD, kD,
      (long long)2 * kD * kD, (long long)kD * 2 * kD);
  k_transpose_cvt<<<dim3(kD / 32, kD / 32, 1), 256, 0, stream>>>(
      bWh1, h1_w, kD, kD, kD, kD, 0, 0);
  k_transpose_cvt<<<dim3(kD / 32, kNPAD / 32, 1), 256, 0, stream>>>(
      bWh2, h2_w, kD, kNPAD, kD, kC * kPL, 0, 0);
  k_pad_bias<<<2, 256, 0, stream>>>(fBias2, h2_b, kNPAD, kC * kPL);

  // ---- embedding ----
  k_prep_patches<<<blocks((long long)kRowsPat * kKP), 256, 0, stream>>>(bApat, patches);
  gemm(stream, G_OUTF, bApat, bWpat, patch_b, nullptr, fPV, nullptr,
       kRowsPat, kKP, kD);
  k_embed<<<kRowsTok, 256, 0, stream>>>(fX, fPV, stats, topo, stat_w, stat_b,
                                        topo_w, topo_b, mask, mask_tok);
  k_ln<<<kRowsTok, 32, 0, stream>>>(fX, emb_g, emb_b, fX, nullptr);

  // ---- transformer layers ----
  for (int l = 0; l < kL; ++l) {
    k_ln<<<kRowsTok, 32, 0, stream>>>(fX, ln1_g + (size_t)l * kD,
                                      ln1_b + (size_t)l * kD, nullptr, bH);
    gemm(stream, G_OUTB, bH, bWqkv + (size_t)l * 3 * kD * kD,
         qkv_b + (size_t)l * 3 * kD, nullptr, nullptr, bQKV,
         kRowsTok, kD, 3 * kD);
    k_attn<<<(kB * kH) / 4, 128, 0, stream>>>(bQKV, bAttn);
    gemm(stream, G_RES_OUTF, bAttn, bWproj + (size_t)l * kD * kD,
         proj_b + (size_t)l * kD, fX, fX, nullptr, kRowsTok, kD, kD);
    k_ln<<<kRowsTok, 32, 0, stream>>>(fX, ln2_g + (size_t)l * kD,
                                      ln2_b + (size_t)l * kD, nullptr, bH);
    gemm(stream, G_RELU_OUTB, bH, bWf1 + (size_t)l * 2 * kD * kD,
         f1_b + (size_t)l * 2 * kD, nullptr, nullptr, bMid,
         kRowsTok, kD, 2 * kD);
    gemm(stream, G_RES_OUTF, bMid, bWf2 + (size_t)l * kD * 2 * kD,
         f2_b + (size_t)l * kD, fX, fX, nullptr, kRowsTok, 2 * kD, kD);
  }

  // ---- reconstruction head + loss ----
  k_gather<<<blocks((long long)kRowsPat * kD), 256, 0, stream>>>(bHeadI, fX);
  gemm(stream, G_RELU_OUTB, bHeadI, bWh1, h1_b, nullptr, nullptr, bHeadM,
       kRowsPat, kD, kD);
  gemm(stream, G_OUTF, bHeadM, bWh2, fBias2, nullptr, fPred, nullptr,
       kRowsPat, kD, kNPAD);
  k_loss<<<kRowsPat, 32, 0, stream>>>(fPred, patches, mask, fAcc);
  k_final<<<1, 1, 0, stream>>>((float*)d_out, fAcc);
}
